// Head_61787399520256
// MI455X (gfx1250) — matrix operand order repaired and checked by reading, not compiled, not run on hardware
//
#include <hip/hip_runtime.h>
#include <hip/hip_bf16.h>

// ---- CDNA5 wave32 WMMA attention: x->QKV (bf16 wmma), fused causal softmax-attn ----

typedef __attribute__((ext_vector_type(16))) __bf16 v16bf;
typedef __attribute__((ext_vector_type(2)))  __bf16 v2bf;
typedef __attribute__((ext_vector_type(8)))  float  v8f;

#define Bdim 256
#define Tdim 256
#define Cdim 384
#define Hdim 64

#if defined(__has_builtin)
#if __has_builtin(__builtin_amdgcn_cvt_pk_bf16_f32)
#define HAVE_CVT_PK_BF16 1
#endif
#endif

// packed f32x2 -> bf16x2 (1 VALU op when hardware cvt is exposed)
static __device__ __forceinline__ v2bf f2bf2(float a, float b) {
#ifdef HAVE_CVT_PK_BF16
  auto t = __builtin_amdgcn_cvt_pk_bf16_f32(a, b);
  return __builtin_bit_cast(v2bf, t);
#else
  unsigned ua = __builtin_bit_cast(unsigned, a);
  unsigned ub = __builtin_bit_cast(unsigned, b);
  ua += 0x7FFFu + ((ua >> 16) & 1u);
  ub += 0x7FFFu + ((ub >> 16) & 1u);
  unsigned p = (ua >> 16) | (ub & 0xFFFF0000u);
  return __builtin_bit_cast(v2bf, p);
#endif
}

static __device__ __forceinline__ __bf16 f2bf(float f) { return f2bf2(f, 0.f)[0]; }

union FragU { uint4 q[2]; __bf16 e[16]; v2bf e2[8]; v16bf v; };

static __device__ __forceinline__ v8f wmma_bf16(v16bf a, v16bf b, v8f c) {
  return __builtin_amdgcn_wmma_f32_16x16x32_bf16(false, a, false, b, (short)0, c,
                                                 false, false);
}

// ------------------------------------------------------------------
// Kernel 0: convert W (f32 [C,H]) -> bf16 [3][C][H] in workspace
// ------------------------------------------------------------------
__global__ void prep_w(const float* __restrict__ Wq, const float* __restrict__ Wk,
                       const float* __restrict__ Wv, __bf16* __restrict__ Wb) {
  int i = blockIdx.x * blockDim.x + threadIdx.x;
  if (i >= 3 * Cdim * Hdim) return;
  int w = i / (Cdim * Hdim), r = i % (Cdim * Hdim);
  const float* src = (w == 0) ? Wq : (w == 1) ? Wk : Wv;
  Wb[i] = f2bf(src[r]);
}

// ------------------------------------------------------------------
// Kernel 1: Q/K/V projections. One wave = 16 rows of x, full H=64.
// Q,V stored [B*T, H] bf16; K stored transposed [B, H, T] bf16.
// Q is pre-scaled by H^-0.5 so the attention kernel skips the multiply.
// ------------------------------------------------------------------
__global__ __launch_bounds__(256) void proj_qkv(const float* __restrict__ x,
                                                const __bf16* __restrict__ Wb,
                                                __bf16* __restrict__ Q,
                                                __bf16* __restrict__ Kt,
                                                __bf16* __restrict__ V) {
  const int wave = threadIdx.x >> 5;
  const int lane = threadIdx.x & 31;
  const int lo = lane & 15, hi = lane >> 4;
  const int row0 = (blockIdx.x * 8 + wave) * 16;

  v8f accQ[4], accK[4], accV[4];
  const v8f z = {0.f, 0.f, 0.f, 0.f, 0.f, 0.f, 0.f, 0.f};
#pragma unroll
  for (int nt = 0; nt < 4; ++nt) { accQ[nt] = z; accK[nt] = z; accV[nt] = z; }

  const float* xrow = x + (size_t)(row0 + lo) * Cdim;

  for (int kc = 0; kc < Cdim; kc += 32) {
    // A fragment: row m = lo ; elements e<8 -> k = kc+8*hi+e, e>=8 -> kc+16+8*hi+(e-8)
    FragU a;
    const float* p0 = xrow + kc + 8 * hi;
    const float* p1 = xrow + kc + 16 + 8 * hi;
    float4 f0 = *(const float4*)(p0);
    float4 f1 = *(const float4*)(p0 + 4);
    float4 f2 = *(const float4*)(p1);
    float4 f3 = *(const float4*)(p1 + 4);
    a.e2[0] = f2bf2(f0.x, f0.y);
    a.e2[1] = f2bf2(f0.z, f0.w);
    a.e2[2] = f2bf2(f1.x, f1.y);
    a.e2[3] = f2bf2(f1.z, f1.w);
    a.e2[4] = f2bf2(f2.x, f2.y);
    a.e2[5] = f2bf2(f2.z, f2.w);
    a.e2[6] = f2bf2(f3.x, f3.y);
    a.e2[7] = f2bf2(f3.z, f3.w);

    // B fragments: lane owns H col (nt*16 + lo); e<8 -> k = kc+8*hi+e, e>=8 -> kc+16+8*hi+(e-8)
    const __bf16* wcol = Wb + (size_t)(kc + 8 * hi) * Hdim + lo;
#pragma unroll
    for (int nt = 0; nt < 4; ++nt) {
      FragU bq, bk, bv;
#pragma unroll
      for (int e = 0; e < 16; ++e) {
        const size_t o = (size_t)((e & 7) + ((e >> 3) << 4)) * Hdim + nt * 16;
        bq.e[e] = wcol[o];
        bk.e[e] = wcol[(size_t)1 * Cdim * Hdim + o];
        bv.e[e] = wcol[(size_t)2 * Cdim * Hdim + o];
      }
      accQ[nt] = wmma_bf16(a.v, bq.v, accQ[nt]);
      accK[nt] = wmma_bf16(a.v, bk.v, accK[nt]);
      accV[nt] = wmma_bf16(a.v, bv.v, accV[nt]);
    }
  }

  // C/D layout: row = r + 8*hi, col = nt*16 + lo
  const int batch = row0 / Tdim;
#pragma unroll
  for (int nt = 0; nt < 4; ++nt) {
#pragma unroll
    for (int r = 0; r < 8; ++r) {
      const int grow = row0 + r + 8 * hi;
      const int h = nt * 16 + lo;
      Q[(size_t)grow * Hdim + h] = f2bf(accQ[nt][r] * 0.125f);  // fold H^-0.5
      V[(size_t)grow * Hdim + h] = f2bf(accV[nt][r]);
      const int tl = grow & (Tdim - 1);
      Kt[((size_t)batch * Hdim + h) * Tdim + tl] = f2bf(accK[nt][r]);
    }
  }
}

// ------------------------------------------------------------------
// Kernel 2: fused causal attention. One wave = 16 query rows.
// S tiles live in registers (16 x v8f); softmax via shfl_xor within
// 16-lane groups; P bounced through wave-private LDS for A fragments.
// ------------------------------------------------------------------
__global__ __launch_bounds__(64) void attn(const __bf16* __restrict__ Q,
                                           const __bf16* __restrict__ Kt,
                                           const __bf16* __restrict__ V,
                                           float* __restrict__ out) {
  __shared__ __align__(16) __bf16 P[2][16][272];  // 272 pad keeps b128 aligned

  const int wave = threadIdx.x >> 5;
  const int lane = threadIdx.x & 31;
  const int lo = lane & 15, hi = lane >> 4;
  const int batch = blockIdx.x >> 3;
  const int t0 = ((blockIdx.x & 7) * 2 + wave) * 16;  // query tile base in batch

  // Q A-fragments for both 32-wide K chunks of H=64
  FragU aq0, aq1;
  const __bf16* qrow = Q + ((size_t)batch * Tdim + t0 + lo) * Hdim;
  aq0.q[0] = *(const uint4*)(qrow + 8 * hi);
  aq0.q[1] = *(const uint4*)(qrow + 16 + 8 * hi);
  aq1.q[0] = *(const uint4*)(qrow + 32 + 8 * hi);
  aq1.q[1] = *(const uint4*)(qrow + 48 + 8 * hi);

  // ---- S = (Q*H^-0.5) K^T with causal mask, all 16 tiles in registers ----
  v8f S[16];
  const __bf16* ktb = Kt + (size_t)batch * Hdim * Tdim;
  const v8f z = {0.f, 0.f, 0.f, 0.f, 0.f, 0.f, 0.f, 0.f};
#pragma unroll
  for (int n = 0; n < 16; ++n) {
    FragU b0, b1;  // lane owns s-col n*16+lo; elements e -> h = 8*hi+(e&7)+16*(e>>3) [+32]
#pragma unroll
    for (int e = 0; e < 16; ++e) {
      const int kh = 8 * hi + (e & 7) + ((e >> 3) << 4);
      b0.e[e] = ktb[(size_t)kh * Tdim + n * 16 + lo];
      b1.e[e] = ktb[(size_t)(32 + kh) * Tdim + n * 16 + lo];
    }
    v8f s = wmma_bf16(aq0.v, b0.v, z);
    s = wmma_bf16(aq1.v, b1.v, s);
    const int scol = n * 16 + lo;
#pragma unroll
    for (int r = 0; r < 8; ++r) {
      const int trow = t0 + r + 8 * hi;
      s[r] = (scol <= trow) ? s[r] : -3.0e38f;  // pure select; scale pre-folded
    }
    S[n] = s;
  }

  // ---- softmax: row stats reduced across the 16 lanes sharing each row ----
  float rmax[8], rsum[8];
#pragma unroll
  for (int r = 0; r < 8; ++r) {
    float m = S[0][r];
#pragma unroll
    for (int n = 1; n < 16; ++n) m = fmaxf(m, S[n][r]);
#pragma unroll
    for (int off = 1; off < 16; off <<= 1) m = fmaxf(m, __shfl_xor(m, off, 32));
    rmax[r] = m;
    rsum[r] = 0.f;
  }
#pragma unroll
  for (int n = 0; n < 16; ++n) {
#pragma unroll
    for (int r = 0; r < 8; r += 2) {
      float e0 = __expf(S[n][r] - rmax[r]);
      float e1 = __expf(S[n][r + 1] - rmax[r + 1]);
      rsum[r] += e0;
      rsum[r + 1] += e1;
      v2bf p = f2bf2(e0, e1);
      P[wave][r + 8 * hi][n * 16 + lo] = p[0];      // wave-private LDS tile
      P[wave][r + 1 + 8 * hi][n * 16 + lo] = p[1];
    }
  }
#pragma unroll
  for (int r = 0; r < 8; ++r) {
    float s = rsum[r];
#pragma unroll
    for (int off = 1; off < 16; off <<= 1) s += __shfl_xor(s, off, 32);
    rsum[r] = s;
  }

  // ---- O = P V ; V's [T,H] layout is exactly the B-fragment layout ----
  v8f O[4];
#pragma unroll
  for (int nt = 0; nt < 4; ++nt) O[nt] = z;
  const __bf16* vb = V + (size_t)batch * Tdim * Hdim;
#pragma unroll
  for (int sc = 0; sc < 8; ++sc) {
    FragU ap;  // A fragment from LDS: row m = lo, elements = s per A mapping
    ap.q[0] = *(const uint4*)(&P[wave][lo][sc * 32 + 8 * hi]);
    ap.q[1] = *(const uint4*)(&P[wave][lo][sc * 32 + 16 + 8 * hi]);
    const int s = sc * 32 + 8 * hi;  // lane's base V row (K index), table order
#pragma unroll
    for (int nt = 0; nt < 4; ++nt) {
      FragU bv;
#pragma unroll
      for (int e = 0; e < 16; ++e)
        bv.e[e] = vb[(size_t)(s + (e & 7) + ((e >> 3) << 4)) * Hdim + nt * 16 + lo];
      O[nt] = wmma_bf16(ap.v, bv.v, O[nt]);
    }
  }

  // ---- normalize + store f32 output ----
  float* ob = out + ((size_t)batch * Tdim + t0) * Hdim;
#pragma unroll
  for (int r = 0; r < 8; ++r) {
    const float inv = 1.0f / rsum[r];
#pragma unroll
    for (int nt = 0; nt < 4; ++nt)
      ob[(size_t)(r + 8 * hi) * Hdim + nt * 16 + lo] = O[nt][r] * inv;
  }
}

// ------------------------------------------------------------------
extern "C" void kernel_launch(void* const* d_in, const int* in_sizes, int n_in,
                              void* d_out, int out_size, void* d_ws, size_t ws_size,
                              hipStream_t stream) {
  const float* x  = (const float*)d_in[0];
  const float* Wq = (const float*)d_in[1];
  const float* Wk = (const float*)d_in[2];
  const float* Wv = (const float*)d_in[3];

  // workspace layout (bytes): Wb bf16 [3][C][H] | Q bf16 [B*T,H] | Kt bf16 [B,H,T] | V bf16 [B*T,H]
  const size_t WB_BYTES  = (size_t)3 * Cdim * Hdim * 2;        // 147456
  const size_t QKV_BYTES = (size_t)Bdim * Tdim * Hdim * 2;     // 8 MiB each
  char* ws = (char*)d_ws;
  __bf16* Wb = (__bf16*)ws;
  __bf16* Qw = (__bf16*)(ws + WB_BYTES);
  __bf16* Kt = (__bf16*)(ws + WB_BYTES + QKV_BYTES);
  __bf16* Vw = (__bf16*)(ws + WB_BYTES + 2 * QKV_BYTES);

  prep_w<<<(3 * Cdim * Hdim + 255) / 256, 256, 0, stream>>>(Wq, Wk, Wv, Wb);
  proj_qkv<<<(Bdim * Tdim) / (16 * 8), 256, 0, stream>>>(x, Wb, Qw, Kt, Vw);
  attn<<<Bdim * (Tdim / 32), 64, 0, stream>>>(Qw, Kt, Vw, (float*)d_out);
}
